// DescriptorModuleSpecies_11854109737449
// MI455X (gfx1250) — matrix operand order repaired and checked by reading, not compiled, not run on hardware
//
#include <hip/hip_runtime.h>

// DescriptorModuleSpecies for MI455X (gfx1250):
// one wave32 per atom; filter MLP + all three einsums on
// v_wmma_f32_16x16x32_f16 (f16 operands, f32 accumulation).
// G / R / R1 kept transposed in LDS so B fragments and G^T A fragments
// are contiguous b128 loads and C stores are single b128 stores.

#define S_SNAP 8
#define N_ATOM 4096
#define M_NEI  64
#define WAVES  4          // waves (atoms) per workgroup -> 128 threads
#define LBOX   20.0f
#define R_CS_  2.0f
#define R_C_   6.0f

typedef __attribute__((ext_vector_type(16))) _Float16 v16h;
typedef __attribute__((ext_vector_type(8)))  _Float16 v8h;
typedef __attribute__((ext_vector_type(4)))  _Float16 v4h;
typedef __attribute__((ext_vector_type(8)))  float    v8f;

__device__ __forceinline__ v8f wmma16(v16h a, v16h b, v8f c) {
    // D = A(16x32 f16) * B(32x16 f16) + C(16x16 f32)
    return __builtin_amdgcn_wmma_f32_16x16x32_f16(
        false, a, false, b, (short)0, c, false, false);
}

// Unconditional 2-half LDS load, then select (forces v_cndmask, not a branch).
__device__ __forceinline__ void load_pair(v16h& dst, int i2, const _Float16* p, bool ok) {
    union { uint32_t u; _Float16 h[2]; } x;
    const uint32_t w = *(const uint32_t*)p;
    x.u = ok ? w : 0u;
    dst[i2] = x.h[0];
    dst[i2 + 1] = x.h[1];
}

// ---- A fragment (16x32 f16), row-major source with zero-padded K ----
// lane L: row M = m0 + (L&15); halves i -> K = (i<8 ? i : 8+i) + 8*(L>=16)
__device__ __forceinline__ v16h load_a_part(const _Float16* src, int ld, int m0,
                                            int Ktot, int lane) {
    v16h a;
    const int m  = m0 + (lane & 15);
    const int kb = (lane >= 16) ? 8 : 0;
    const _Float16* row = src + m * ld;
#pragma unroll
    for (int u = 0; u < 8; ++u) {
        const int kp = ((u < 4) ? (2 * u) : (16 + 2 * (u - 4))) + kb;
        load_pair(a, 2 * u, row + kp, kp < Ktot);
    }
    return a;
}

// A fragment, row-major source with ld == 64 and full K: two b128 loads
__device__ __forceinline__ v16h load_a_full(const _Float16* src, int m0, int k0,
                                            int lane) {
    const int m  = m0 + (lane & 15);
    const int kb = (lane >= 16) ? 8 : 0;
    const _Float16* row = src + m * 64 + k0 + kb;
    const v8h lo = *(const v8h*)(row);        // K = kb .. kb+7
    const v8h hi = *(const v8h*)(row + 16);   // K = 16+kb .. 16+kb+7
    v16h a;
#pragma unroll
    for (int j = 0; j < 8; ++j) { a[j] = lo[j]; a[8 + j] = hi[j]; }
    return a;
}

// ---- B fragment (32x16 f16) from column-major (transposed) source ----
// lane L: col N = n0 + (L&15); halves i -> K = (i<8 ? i : 8+i) + 8*(L>=16)
__device__ __forceinline__ v16h load_b_cm_full(const _Float16* srcT, int k0,
                                               int lane) {
    const int n  = lane & 15;
    const int kb = (lane >= 16) ? 8 : 0;
    const _Float16* col = srcT + n * 64 + k0 + kb;
    const v8h lo = *(const v8h*)(col);
    const v8h hi = *(const v8h*)(col + 16);
    v16h b;
#pragma unroll
    for (int j = 0; j < 8; ++j) { b[j] = lo[j]; b[8 + j] = hi[j]; }
    return b;
}

__device__ __forceinline__ v16h load_b_cm_part(const _Float16* srcT, int ldm,
                                               int n0, int Ktot, int Ntot,
                                               int lane) {
    v16h b;
    const int n   = n0 + (lane & 15);
    const bool no = (n < Ntot);
    const int kb  = (lane >= 16) ? 8 : 0;
    const _Float16* col = srcT + (no ? n : n0) * ldm;
#pragma unroll
    for (int u = 0; u < 8; ++u) {
        const int k = ((u < 4) ? (2 * u) : (16 + 2 * (u - 4))) + kb;
        load_pair(b, 2 * u, col + k, no && (k < Ktot));
    }
    return b;
}

// ---- C fragment stores ----
// column-major dest: lane owns col n0+(L&15), rows m0+mb .. +7 -> one b128
__device__ __forceinline__ void store_c_cm(_Float16* dstT, int ldm, int m0,
                                           int n0, float bv, v8f c, int lane) {
    const int n  = n0 + (lane & 15);
    const int mb = (lane >= 16) ? 8 : 0;
    v8h h;
#pragma unroll
    for (int j = 0; j < 8; ++j) h[j] = (_Float16)(c[j] + bv);
    *(v8h*)(dstT + n * ldm + m0 + mb) = h;
}

// row-major dest with bias+relu (h1/h2): 8 strided b16 stores
__device__ __forceinline__ void store_c_rm(_Float16* dst, int ld, int m0,
                                           int Ntot, float bv, bool relu,
                                           v8f c, int lane) {
    const int ng = lane & 15;
    if (ng < Ntot) {
        const int mb = m0 + ((lane >= 16) ? 8 : 0);
#pragma unroll
        for (int j = 0; j < 8; ++j) {
            float v = c[j] + bv;
            if (relu) v = v > 0.0f ? v : 0.0f;
            dst[(mb + j) * ld + ng] = (_Float16)v;
        }
    }
}

__global__ __launch_bounds__(32 * WAVES)
void descriptor_wmma_kernel(const float* __restrict__ inputs,
                            const int*   __restrict__ types,
                            const int*   __restrict__ neigh,
                            const float* __restrict__ es1_w, const float* __restrict__ es1_b,
                            const float* __restrict__ es2_w, const float* __restrict__ es2_b,
                            const float* __restrict__ fs1_w, const float* __restrict__ fs1_b,
                            const float* __restrict__ fs2_w, const float* __restrict__ fs2_b,
                            const float* __restrict__ en1_w, const float* __restrict__ en1_b,
                            const float* __restrict__ en2_w, const float* __restrict__ en2_b,
                            const float* __restrict__ en3_w, const float* __restrict__ en3_b,
                            float* __restrict__ out) {
    // ---- shared staging (weights transposed: [n][k]) ----
    __shared__ __align__(16) _Float16 sW1T[8 * 4];
    __shared__ __align__(16) _Float16 sW2T[16 * 8];
    __shared__ __align__(16) _Float16 sW3T[32 * 16];
    __shared__ float sB1[8], sB2[16], sB3[32];
    __shared__ float sTD[4 * 4];   // type-pair embedding table

    __shared__ __align__(16) _Float16 sST [WAVES][64 * 4];   // s_tilde   (row major)
    __shared__ __align__(16) _Float16 sRT [WAVES][64 * 4];   // r_tilde   (row major)
    __shared__ __align__(16) _Float16 sRTT[WAVES][4 * 64];   // r_tilde^T
    __shared__ __align__(16) _Float16 sH1 [WAVES][64 * 8];   // row major
    __shared__ __align__(16) _Float16 sH2 [WAVES][64 * 16];  // row major
    __shared__ __align__(16) _Float16 sGT [WAVES][32 * 64];  // G^T: [g][m]
    __shared__ __align__(16) _Float16 sRmT[WAVES][16 * 16];  // R^T-ish: [n][d]
    __shared__ __align__(16) _Float16 sR1T[WAVES][16 * 64];  // R1^T: [n][m]

    const int tid  = threadIdx.x;
    const int wid  = tid >> 5;
    const int lane = tid & 31;

    // stage filter-net weights transposed (f32 -> f16) and biases
    for (int i = tid; i < 32;  i += blockDim.x) { const int n = i >> 2, k = i & 3;  sW1T[i] = (_Float16)en1_w[k * 8  + n]; }
    for (int i = tid; i < 128; i += blockDim.x) { const int n = i >> 3, k = i & 7;  sW2T[i] = (_Float16)en2_w[k * 16 + n]; }
    for (int i = tid; i < 512; i += blockDim.x) { const int n = i >> 4, k = i & 15; sW3T[i] = (_Float16)en3_w[k * 32 + n]; }
    for (int i = tid; i < 8;  i += blockDim.x) sB1[i] = en1_b[i];
    for (int i = tid; i < 16; i += blockDim.x) sB2[i] = en2_b[i];
    for (int i = tid; i < 32; i += blockDim.x) sB3[i] = en3_b[i];

    // type-pair table: types in {0,1} -> the whole es/fs net has 4 inputs
    if (tid < 4) {
        const float ta = (float)(tid >> 1), tb = (float)(tid & 1);
        float e[4] = {0.f, 0.f, 0.f, 0.f};
#pragma unroll
        for (int dir = 0; dir < 2; ++dir) {
            const float x0 = dir ? tb : ta;
            const float x1 = dir ? ta : tb;
            float h[4];
#pragma unroll
            for (int j = 0; j < 4; ++j) {
                const float v = x0 * es1_w[j] + x1 * es1_w[4 + j] + es1_b[j];
                h[j] = v > 0.f ? v : 0.f;
            }
#pragma unroll
            for (int j = 0; j < 4; ++j) {
                float v = es2_b[j];
#pragma unroll
                for (int k = 0; k < 4; ++k) v += h[k] * es2_w[k * 4 + j];
                e[j] += v;
            }
        }
        float f[4];
#pragma unroll
        for (int j = 0; j < 4; ++j) {
            float v = fs1_b[j];
#pragma unroll
            for (int k = 0; k < 4; ++k) v += e[k] * fs1_w[k * 4 + j];
            f[j] = v > 0.f ? v : 0.f;
        }
#pragma unroll
        for (int j = 0; j < 4; ++j) {
            float v = fs2_b[j];
#pragma unroll
            for (int k = 0; k < 4; ++k) v += f[k] * fs2_w[k * 4 + j];
            sTD[tid * 4 + j] = v;
        }
    }
    __syncthreads();

    // ---- per-wave (per-atom) work ----
    const int atomIdx = blockIdx.x * WAVES + wid;
    const int s = atomIdx >> 12;               // N_ATOM = 4096
    _Float16* ST  = sST[wid];
    _Float16* RT  = sRT[wid];
    _Float16* RTT = sRTT[wid];
    _Float16* H1  = sH1[wid];
    _Float16* H2  = sH2[wid];
    _Float16* GT  = sGT[wid];
    _Float16* RmT = sRmT[wid];
    _Float16* R1T = sR1T[wid];

    // geometry + cutoff + embedding lookup -> r_tilde, r_tilde^T, s_tilde
    {
        const float pix = inputs[(size_t)atomIdx * 3 + 0];
        const float piy = inputs[(size_t)atomIdx * 3 + 1];
        const float piz = inputs[(size_t)atomIdx * 3 + 2];
        const int   ti  = types[atomIdx];
#pragma unroll
        for (int half = 0; half < 2; ++half) {
            const int m = lane + half * 32;
            const int j = neigh[(size_t)atomIdx * M_NEI + m];
            const bool msk = (j < 0);
            const int jj = msk ? 0 : j;
            const float* pj = inputs + (size_t)(s * N_ATOM + jj) * 3;
            float dx = pj[0] - pix;
            float dy = pj[1] - piy;
            float dz = pj[2] - piz;
            dx -= LBOX * rintf(dx * (1.0f / LBOX));   // minimum image (RNE = jnp.round)
            dy -= LBOX * rintf(dy * (1.0f / LBOX));
            dz -= LBOX * rintf(dz * (1.0f / LBOX));
            const float r = sqrtf(dx * dx + dy * dy + dz * dz + 1e-12f);
            const float u = (r - R_CS_) / (R_C_ - R_CS_);
            const float cw = 0.5f * __cosf(3.14159265358979f * u) + 0.5f;
            const float sw = (r < R_CS_) ? 1.0f : ((r < R_C_) ? cw : 0.0f);
            const float sv = msk ? 0.0f : sw / r;
            const float ir = sv / r;
            v4h rt;
            rt[0] = (_Float16)sv;
            rt[1] = (_Float16)(dx * ir);
            rt[2] = (_Float16)(dy * ir);
            rt[3] = (_Float16)(dz * ir);
            *(v4h*)(RT + m * 4) = rt;                 // one b64 store
            RTT[0 * 64 + m] = rt[0];
            RTT[1 * 64 + m] = rt[1];
            RTT[2 * 64 + m] = rt[2];
            RTT[3 * 64 + m] = rt[3];
            const int tj = types[s * N_ATOM + jj];
            const float* td = &sTD[(ti * 2 + tj) * 4];
            v4h st;
            st[0] = (_Float16)(td[0] * sv);
            st[1] = (_Float16)(td[1] * sv);
            st[2] = (_Float16)(td[2] * sv);
            st[3] = (_Float16)(td[3] * sv);
            *(v4h*)(ST + m * 4) = st;                 // one b64 store
        }
    }
    // all LDS producer/consumer traffic below is intra-wave; DS ops from one
    // wave complete in order, so no barrier is needed.

    const int nl = lane & 15;
    const float b1raw = sB1[nl & 7];
    const float bv1   = (nl < 8) ? b1raw : 0.0f;
    const float bv2   = sB2[nl];
    const float bv3a  = sB3[nl];
    const float bv3b  = sB3[16 + nl];

    // ---- filter MLP: 64x4 -> 64x8 -> 64x16 -> 64x32 (G stored transposed) ----
    {
        const v16h bW1 = load_b_cm_part(sW1T, 4, 0, 4, 8, lane);
#pragma unroll
        for (int t = 0; t < 4; ++t) {
            v8f c = {};
            c = wmma16(load_a_part(ST, 4, t * 16, 4, lane), bW1, c);
            store_c_rm(H1, 8, t * 16, 8, bv1, /*relu=*/true, c, lane);
        }
        const v16h bW2 = load_b_cm_part(sW2T, 8, 0, 8, 16, lane);
#pragma unroll
        for (int t = 0; t < 4; ++t) {
            v8f c = {};
            c = wmma16(load_a_part(H1, 8, t * 16, 8, lane), bW2, c);
            store_c_rm(H2, 16, t * 16, 16, bv2, /*relu=*/true, c, lane);
        }
        const v16h bW3a = load_b_cm_part(sW3T, 16, 0,  16, 32, lane);
        const v16h bW3b = load_b_cm_part(sW3T, 16, 16, 16, 32, lane);
#pragma unroll
        for (int t = 0; t < 4; ++t) {
            const v16h a = load_a_part(H2, 16, t * 16, 16, lane);
            v8f c0 = {};
            c0 = wmma16(a, bW3a, c0);
            store_c_cm(GT, 64, t * 16, 0, bv3a, c0, lane);   // cols 0..15 of G
            v8f c1 = {};
            c1 = wmma16(a, bW3b, c1);
            store_c_cm(GT, 64, t * 16, 16, bv3b, c1, lane);  // cols 16..31
        }
    }

    // ---- R[4x16] = r_tilde^T (4x64) @ G[:, :16] ----
    {
        v8f c = {};
        c = wmma16(load_a_full(RTT, 0, 0,  lane), load_b_cm_full(GT, 0,  lane), c);
        c = wmma16(load_a_full(RTT, 0, 32, lane), load_b_cm_full(GT, 32, lane), c);
        store_c_cm(RmT, 16, 0, 0, 0.0f, c, lane);   // rows d>=4 garbage, never read
    }

    // ---- R1[64x16] = r_tilde (64x4) @ R (4x16), stored transposed ----
    {
        const v16h bR = load_b_cm_part(RmT, 16, 0, 4, 16, lane);
#pragma unroll
        for (int t = 0; t < 4; ++t) {
            v8f c = {};
            c = wmma16(load_a_part(RT, 4, t * 16, 4, lane), bR, c);
            store_c_cm(R1T, 64, t * 16, 0, 0.0f, c, lane);
        }
    }

    // ---- D[32x16] = G^T (32x64) @ R1 (64x16) -> global out ----
    {
        const v16h bD0 = load_b_cm_full(R1T, 0,  lane);
        const v16h bD1 = load_b_cm_full(R1T, 32, lane);
#pragma unroll
        for (int t = 0; t < 2; ++t) {
            v8f c = {};
            c = wmma16(load_a_full(GT, t * 16, 0,  lane), bD0, c);
            c = wmma16(load_a_full(GT, t * 16, 32, lane), bD1, c);
            const int gb = t * 16 + ((lane >= 16) ? 8 : 0);
#pragma unroll
            for (int j = 0; j < 8; ++j) {
                // lanes 0..15 cover one contiguous 64B row, lanes 16..31 another
                out[((size_t)atomIdx * 32 + (gb + j)) * 16 + nl] = c[j];
            }
        }
    }
}

extern "C" void kernel_launch(void* const* d_in, const int* in_sizes, int n_in,
                              void* d_out, int out_size, void* d_ws, size_t ws_size,
                              hipStream_t stream) {
    const float* inputs = (const float*)d_in[0];
    const int*   types  = (const int*)d_in[1];
    const int*   neigh  = (const int*)d_in[2];
    const float* es1_w  = (const float*)d_in[3];
    const float* es1_b  = (const float*)d_in[4];
    const float* es2_w  = (const float*)d_in[5];
    const float* es2_b  = (const float*)d_in[6];
    const float* fs1_w  = (const float*)d_in[7];
    const float* fs1_b  = (const float*)d_in[8];
    const float* fs2_w  = (const float*)d_in[9];
    const float* fs2_b  = (const float*)d_in[10];
    const float* en1_w  = (const float*)d_in[11];
    const float* en1_b  = (const float*)d_in[12];
    const float* en2_w  = (const float*)d_in[13];
    const float* en2_b  = (const float*)d_in[14];
    const float* en3_w  = (const float*)d_in[15];
    const float* en3_b  = (const float*)d_in[16];

    const int totalAtoms = S_SNAP * N_ATOM;          // 32768
    dim3 grid(totalAtoms / WAVES);                   // 8192 workgroups
    dim3 block(32 * WAVES);                          // 4 waves -> 4 atoms per WG
    descriptor_wmma_kernel<<<grid, block, 0, stream>>>(
        inputs, types, neigh,
        es1_w, es1_b, es2_w, es2_b, fs1_w, fs1_b, fs2_w, fs2_b,
        en1_w, en1_b, en2_w, en2_b, en3_w, en3_b,
        (float*)d_out);
}
